// GCN_28295244546111
// MI455X (gfx1250) — matrix operand order repaired and checked by reading, not compiled, not run on hardware
//
#include <hip/hip_runtime.h>

typedef __attribute__((ext_vector_type(16))) _Float16 v16h;
typedef __attribute__((ext_vector_type(8)))  float    v8f;

#define D_IN  64
#define H_MID 128
#define O_OUT 32

// EPS=1: exp(1)=2.718281828459045
// scale = (e+1)/(e-1), bias = -1/(e-1) + ALPHA(0.5)
#define PRIV_A 2.1639534137f
#define PRIV_B (-0.0819767068f)

// ---------------- init / norm kernels ----------------

__global__ void k_fill(float* __restrict__ p, int n, float v) {
  int i = blockIdx.x * blockDim.x + threadIdx.x;
  if (i < n) p[i] = v;
}

__global__ void k_out_bias(float* __restrict__ out, const float* __restrict__ b2, int n) {
  int i = blockIdx.x * blockDim.x + threadIdx.x;
  if (i < n) out[i] = b2[i & (O_OUT - 1)];
}

__global__ void k_count(const int* __restrict__ row, float* __restrict__ deg, int E) {
  int e = blockIdx.x * blockDim.x + threadIdx.x;
  if (e < E) atomicAdd(&deg[row[e]], 1.0f);
}

__global__ void k_rsqrt(float* __restrict__ deg, int n) {
  int i = blockIdx.x * blockDim.x + threadIdx.x;
  if (i < n) deg[i] = rsqrtf(deg[i]);
}

// ---------------- W pre-pack into WMMA B-fragment order ----------------
// B 32x16 f16 layout (wave32): lane%16 = N column, half j -> K = (j&7) + 16*(j>>3) + 8*(lane/16).
// W1: frag f = ntile*2 + kstep (8 ntiles, 2 ksteps). W2: frag f = ntile*4 + kstep (2,4).
__global__ void k_pack(const float* __restrict__ W1, const float* __restrict__ W2,
                       _Float16* __restrict__ W1p, _Float16* __restrict__ W2p) {
  int gid = blockIdx.x * blockDim.x + threadIdx.x;
  if (gid < 16 * 512) {                       // W1: 16 frags * 32 lanes * 16 halves
    int f = gid >> 9, rem = gid & 511;
    int lane = rem >> 4, j = rem & 15;
    int ntile = f >> 1, kstep = f & 1;
    int K = (j & 7) + ((j >> 3) << 4) + ((lane >> 4) << 3) + kstep * 32;
    int C = (lane & 15) + ntile * 16;
    W1p[gid] = (_Float16)W1[K * H_MID + C];
  } else if (gid < 16 * 512 + 8 * 512) {      // W2: 8 frags
    int g2 = gid - 16 * 512;
    int f = g2 >> 9, rem = g2 & 511;
    int lane = rem >> 4, j = rem & 15;
    int ntile = f >> 2, kstep = f & 3;
    int K = (j & 7) + ((j >> 3) << 4) + ((lane >> 4) << 3) + kstep * 32;
    int C = (lane & 15) + ntile * 16;
    W2p[g2] = (_Float16)W2[K * O_OUT + C];
  }
}

// ---------------- SpMM (scatter-add with L2 fp32 atomics) ----------------
// edge e in [0,E): (row[e], col[e]); e in [E, E+N): self loop (e-E, e-E).

__global__ void k_spmm1(const float* __restrict__ x,
                        const int* __restrict__ row, const int* __restrict__ col,
                        const unsigned char* __restrict__ pm,
                        const float* __restrict__ dis,
                        float* __restrict__ h, int E, int M) {
  int gid = blockIdx.x * blockDim.x + threadIdx.x;
  int e = gid >> 6, d = gid & 63;
  if (e >= M) return;
  int r, c;
  if (e < E) { r = row[e]; c = col[e]; } else { r = e - E; c = r; }
  float nr = dis[r] * dis[c];
  float xv = x[r * D_IN + d];
  float v  = pm[r] ? (PRIV_A * xv + PRIV_B) : xv;
  atomicAdd(&h[c * D_IN + d], nr * v);
}

__global__ void k_spmm2(const float* __restrict__ g,
                        const int* __restrict__ row, const int* __restrict__ col,
                        const float* __restrict__ dis,
                        float* __restrict__ out, int E, int M) {
  int gid = blockIdx.x * blockDim.x + threadIdx.x;
  int e = gid >> 5, d = gid & 31;
  if (e >= M) return;
  int r, c;
  if (e < E) { r = row[e]; c = col[e]; } else { r = e - E; c = r; }
  float nr = dis[r] * dis[c];
  atomicAdd(&out[c * O_OUT + d], -nr * g[r * O_OUT + d]);
}

// ---------------- WMMA GEMMs ----------------
// A 16x32 f16 layout (wave32): lane&15 = M row; per-lane halves are two
// contiguous 8-elem K runs: start = ks*32 + (lane&16 ? 8 : 0), second at +16.
__device__ __forceinline__ v16h load_a_frag(const float* __restrict__ S,
                                            int rowStride, int m_base, int ks, int lane) {
  const float* base = S + (m_base + (lane & 15)) * rowStride + ks * 32 + ((lane & 16) ? 8 : 0);
  float4 f0 = ((const float4*)base)[0];
  float4 f1 = ((const float4*)base)[1];
  float4 f2 = ((const float4*)(base + 16))[0];
  float4 f3 = ((const float4*)(base + 16))[1];
  v16h a;
  a[0]=(_Float16)f0.x;  a[1]=(_Float16)f0.y;  a[2]=(_Float16)f0.z;  a[3]=(_Float16)f0.w;
  a[4]=(_Float16)f1.x;  a[5]=(_Float16)f1.y;  a[6]=(_Float16)f1.z;  a[7]=(_Float16)f1.w;
  a[8]=(_Float16)f2.x;  a[9]=(_Float16)f2.y;  a[10]=(_Float16)f2.z; a[11]=(_Float16)f2.w;
  a[12]=(_Float16)f3.x; a[13]=(_Float16)f3.y; a[14]=(_Float16)f3.z; a[15]=(_Float16)f3.w;
  return a;
}

// h1 = relu(h @ W1 + b1): one wave computes a 16x128 row strip.
__global__ void k_gemm1(const float* __restrict__ h, const _Float16* __restrict__ W1p,
                        const float* __restrict__ b1, float* __restrict__ h1, int ntiles) {
  int lane = threadIdx.x & 31;
  int tile = blockIdx.x * 8 + (threadIdx.x >> 5);
  if (tile >= ntiles) return;                  // uniform per wave
  int m_base = tile * 16;
  v8f acc[8] = {};
  const v16h* BP = (const v16h*)W1p;
#pragma unroll
  for (int ks = 0; ks < 2; ++ks) {
    v16h a = load_a_frag(h, D_IN, m_base, ks, lane);
#pragma unroll
    for (int t = 0; t < 8; ++t) {
      v16h b = BP[(t * 2 + ks) * 32 + lane];
      acc[t] = __builtin_amdgcn_wmma_f32_16x16x32_f16(false, a, false, b,
                                                      (short)0, acc[t], false, false);
    }
  }
  // C/D layout: VGPR i -> row m_base + i (+8 for lanes 16-31), col = lane&15.
  int mlo = m_base + ((lane & 16) ? 8 : 0);
  int col0 = lane & 15;
#pragma unroll
  for (int t = 0; t < 8; ++t) {
    int col = t * 16 + col0;
    float bb = b1[col];
#pragma unroll
    for (int i = 0; i < 8; ++i) {
      float v = acc[t][i] + bb;
      h1[(mlo + i) * H_MID + col] = v > 0.0f ? v : 0.0f;
    }
  }
}

// g = h1 @ W2 : 16x32 output strip per wave, K=128 (4 k-steps).
__global__ void k_gemm2(const float* __restrict__ h1, const _Float16* __restrict__ W2p,
                        float* __restrict__ g, int ntiles) {
  int lane = threadIdx.x & 31;
  int tile = blockIdx.x * 8 + (threadIdx.x >> 5);
  if (tile >= ntiles) return;
  int m_base = tile * 16;
  v8f acc[2] = {};
  const v16h* BP = (const v16h*)W2p;
#pragma unroll
  for (int ks = 0; ks < 4; ++ks) {
    v16h a = load_a_frag(h1, H_MID, m_base, ks, lane);
#pragma unroll
    for (int t = 0; t < 2; ++t) {
      v16h b = BP[(t * 4 + ks) * 32 + lane];
      acc[t] = __builtin_amdgcn_wmma_f32_16x16x32_f16(false, a, false, b,
                                                      (short)0, acc[t], false, false);
    }
  }
  int mlo = m_base + ((lane & 16) ? 8 : 0);
  int col0 = lane & 15;
#pragma unroll
  for (int t = 0; t < 2; ++t) {
    int col = t * 16 + col0;
#pragma unroll
    for (int i = 0; i < 8; ++i)
      g[(mlo + i) * O_OUT + col] = acc[t][i];
  }
}

// ---------------- launcher ----------------

static inline size_t align256(size_t v) { return (v + 255) & ~(size_t)255; }

extern "C" void kernel_launch(void* const* d_in, const int* in_sizes, int n_in,
                              void* d_out, int out_size, void* d_ws, size_t ws_size,
                              hipStream_t stream) {
  const float*         x    = (const float*)d_in[0];
  const int*           edge = (const int*)d_in[1];          // [2,E] row-major
  const unsigned char* pm   = (const unsigned char*)d_in[2]; // bool mask, 1 byte
  const float*         W1   = (const float*)d_in[3];
  const float*         b1   = (const float*)d_in[4];
  const float*         W2   = (const float*)d_in[5];
  const float*         b2   = (const float*)d_in[6];
  float*               out  = (float*)d_out;

  const int N = in_sizes[0] / D_IN;
  const int E = in_sizes[1] / 2;
  const int M = E + N;                     // edges + self loops
  const int* row = edge;
  const int* col = edge + E;

  // workspace layout
  char* ws = (char*)d_ws;
  float* dis = (float*)ws;                 ws += align256((size_t)N * 4);
  float* h   = (float*)ws;                 ws += align256((size_t)N * D_IN * 4);
  float* h1  = (float*)ws;                 ws += align256((size_t)N * H_MID * 4);
  float* g   = (float*)ws;                 ws += align256((size_t)N * O_OUT * 4);
  _Float16* W1p = (_Float16*)ws;           ws += align256((size_t)16 * 512 * 2);
  _Float16* W2p = (_Float16*)ws;           ws += align256((size_t)8 * 512 * 2);

  const int B = 256;
  // re-init every call (graph replay safe)
  k_fill<<<(N + B - 1) / B, B, 0, stream>>>(dis, N, 1.0f);          // deg starts at 1 (self loop)
  k_fill<<<(N * D_IN + B - 1) / B, B, 0, stream>>>(h, N * D_IN, 0.0f);
  k_out_bias<<<(N * O_OUT + B - 1) / B, B, 0, stream>>>(out, b2, N * O_OUT);

  k_count<<<(E + B - 1) / B, B, 0, stream>>>(row, dis, E);
  k_rsqrt<<<(N + B - 1) / B, B, 0, stream>>>(dis, N);
  k_pack<<<(16 * 512 + 8 * 512 + B - 1) / B, B, 0, stream>>>(W1, W2, W1p, W2p);

  k_spmm1<<<(M * D_IN + B - 1) / B, B, 0, stream>>>(x, row, col, pm, dis, h, E, M);

  const int ntiles = (N + 15) / 16;        // N=50000 -> 3125, exact
  k_gemm1<<<(ntiles + 7) / 8, B, 0, stream>>>(h, W1p, b1, h1, ntiles);
  k_gemm2<<<(ntiles + 7) / 8, B, 0, stream>>>(h1, W2p, g, ntiles);

  k_spmm2<<<(M * O_OUT + B - 1) / B, B, 0, stream>>>(g, row, col, dis, out, E, M);
}
